// GaussianUpsampling_27264452395638
// MI455X (gfx1250) — matrix operand order repaired and checked by reading, not compiled, not run on hardware
//
#include <hip/hip_runtime.h>
#include <stdint.h>

// ---------------------------------------------------------------------------
// GaussianUpsampling on MI455X (gfx1250, wave32, WMMA + pipelined async LDS)
//   out[b,f,d] = sum_t softmax_t(-0.1*(f - c[b,t])^2) * hs[b,t,d]
// GEMM via v_wmma_f32_16x16x32_f16 with f32 accumulate. H is pre-transposed
// to f16 (one-time kernel) so the hot loop stages it with
// global_load_async_to_lds_b128, double-buffered one iteration ahead so the
// DMA hides under the WMMA phase. B fragments are register-double-buffered
// so ds_load latency hides under the previous WMMA. Softmax math in fp32,
// branchless. Output written with non-temporal stores (single-use 100 MB
// stream) so the L2 keeps the 64x-reused hsT resident.
// ---------------------------------------------------------------------------

#define B_    16
#define TT    512     // T_TEXT
#define AD    384     // ADIM
#define TF    4096    // T_FEATS
#define DELTA 0.1f

#define FBLK  64      // F rows per block
#define KB    32      // K step (WMMA K for f16)
#define NKIT  (TT / KB)
#define PH_LD 40      // padded LDS stride (halfs) for P tile rows
#define HS_LD 40      // padded LDS stride (halfs) for transposed H rows

typedef __attribute__((ext_vector_type(16))) _Float16 v16h;
typedef __attribute__((ext_vector_type(8)))  _Float16 v8h;
typedef __attribute__((ext_vector_type(8)))  float    v8f;

// ---------------------------------------------------------------------------
// Kernel 1: centers, c[b,t] = inclusive_cumsum(ds[b])[t] - 0.5*ds[b,t]
// ---------------------------------------------------------------------------
__global__ __launch_bounds__(TT) void gu_centers(const float* __restrict__ ds,
                                                 float* __restrict__ c) {
  __shared__ float s[TT];
  const int b = blockIdx.x;
  const int t = threadIdx.x;
  const float d = ds[b * TT + t];
  s[t] = d;
  __syncthreads();
#pragma unroll
  for (int off = 1; off < TT; off <<= 1) {
    float v = (t >= off) ? s[t - off] : 0.0f;
    __syncthreads();
    s[t] += v;
    __syncthreads();
  }
  c[b * TT + t] = s[t] - 0.5f * d;
}

// ---------------------------------------------------------------------------
// Kernel 2: transpose+convert: hsT[b][n][k] = (f16) hs[b][k][n]
// fp32 source is read once -> non-temporal loads.
// ---------------------------------------------------------------------------
__global__ __launch_bounds__(256) void gu_transpose(const float* __restrict__ hs,
                                                    _Float16* __restrict__ hsT) {
  __shared__ _Float16 tile[64][72];   // padded
  const int b  = blockIdx.z;
  const int k0 = blockIdx.x * 64;     // TT/64 = 8
  const int n0 = blockIdx.y * 64;     // AD/64 = 6
  const int tx = threadIdx.x & 63;
  const int ty = threadIdx.x >> 6;    // 0..3

  const float* src = hs + ((size_t)b * TT + k0) * AD + n0;
#pragma unroll
  for (int kk = ty; kk < 64; kk += 4)
    tile[kk][tx] = (_Float16)__builtin_nontemporal_load(&src[(size_t)kk * AD + tx]);
  __syncthreads();

  _Float16* dst = hsT + ((size_t)b * AD + n0) * TT + k0;
#pragma unroll
  for (int nn = ty; nn < 64; nn += 4)
    dst[(size_t)nn * TT + tx] = tile[tx][nn];
}

// ---------------------------------------------------------------------------
// Async stage of one K-slice of H^T into LDS (6 x b128 per thread, ASYNCcnt)
// ---------------------------------------------------------------------------
__device__ __forceinline__ void stage_hst_async(const _Float16* __restrict__ hTb,
                                                int k0, _Float16* dstbase, int j) {
#pragma unroll
  for (int g = 0; g < 6; ++g) {
    int c    = g * 256 + j;
    int n    = c >> 2;
    int part = c & 3;
    const void* gp = (const void*)(hTb + (size_t)n * TT + k0 + part * 8);
    unsigned lds_off =
        (unsigned)(uintptr_t)(dstbase + n * HS_LD + part * 8);  // addr[31:0] = LDS offset
    asm volatile("global_load_async_to_lds_b128 %0, %1, off"
                 :: "v"(lds_off), "v"(gp) : "memory");
  }
}

// B fragment load (32x16 f16, ISA layout): lane = col; elems 0-7 <- K kb.., 8-15 <- K kb+16.., kb = 8 for upper half
__device__ __forceinline__ v16h load_bfrag(const _Float16* __restrict__ Hcur,
                                           int wn, int nt, int lane) {
  int ncol = wn * 192 + nt * 16 + (lane & 15);
  const _Float16* pb = Hcur + ncol * HS_LD + (lane >> 4) * 8;
  v8h b0 = *(const v8h*)pb;
  v8h b1 = *(const v8h*)(pb + 16);
  return __builtin_shufflevector(b0, b1, 0,1,2,3,4,5,6,7,
                                         8,9,10,11,12,13,14,15);
}

// ---------------------------------------------------------------------------
// Kernel 3: main GEMM. Block = 256 threads (8 waves): 64 F-rows x 384 cols
// per batch. Waves 4(M) x 2(N); each wave 16 rows x 192 cols = 12 C tiles.
// ---------------------------------------------------------------------------
__global__ __launch_bounds__(256) void gu_main(
    const _Float16* __restrict__ hsT,          // (B, AD, TT) f16, K-contig
    const float* __restrict__ cws,             // (B, TT) centers
    const unsigned char* __restrict__ hmask,   // (B, TF) bool
    const unsigned char* __restrict__ dmask,   // (B, TT) bool
    float* __restrict__ out)                   // (B, TF, AD) f32
{
  __shared__ float c_s[TT];
  __shared__ float dmf_s[TT];      // d_mask as 0.0/1.0
  __shared__ float tval_s[FBLK];
  __shared__ float red_s[FBLK * 4];
  __shared__ float mrow[FBLK];
  __shared__ float sinv[FBLK];
  alignas(16) __shared__ _Float16 Ph[FBLK * PH_LD];       // P tile [r][k]
  alignas(16) __shared__ _Float16 Hst[2][AD * HS_LD];     // H^T tiles, 2 buffers

  const int b    = blockIdx.y;
  const int fblk = blockIdx.x * FBLK;
  const int j    = threadIdx.x;
  const int lane = j & 31;
  const int wave = j >> 5;
  const int wm   = wave >> 1;   // 0..3
  const int wn   = wave & 1;    // 0..1

  const _Float16* hTb = hsT + (size_t)b * AD * TT;

  // ---- prologue: kick off DMA of the first H slice; it overlaps the stats --
  stage_hst_async(hTb, 0, &Hst[0][0], j);

  // ---- stage centers / mask / masked frame index ---------------------------
  for (int t = j; t < TT; t += 256) {
    c_s[t]   = cws[b * TT + t];
    dmf_s[t] = dmask[b * TT + t] ? 1.0f : 0.0f;
  }
  if (j < FBLK) {
    int f = fblk + j;
    tval_s[j] = hmask[b * TF + f] ? (float)f : 0.0f;
  }
  __syncthreads();

  // ---- softmax row stats: branchless, 4 threads per row --------------------
  const int r = j >> 2;
  const int q = j & 3;
  {
    float tv = tval_s[r];
    float lmax = -3.0e38f;
    for (int t = q * 128; t < q * 128 + 128; ++t) {
      float d = tv - c_s[t];
      float e = -DELTA * d * d;
      lmax = fmaxf(lmax, (dmf_s[t] != 0.0f) ? e : -3.0e38f);
    }
    red_s[r * 4 + q] = lmax;
  }
  __syncthreads();
  if (j < FBLK) {
    mrow[j] = fmaxf(fmaxf(red_s[j * 4 + 0], red_s[j * 4 + 1]),
                    fmaxf(red_s[j * 4 + 2], red_s[j * 4 + 3]));
  }
  __syncthreads();
  {
    float tv = tval_s[r], m = mrow[r];
    float ls = 0.0f;
    for (int t = q * 128; t < q * 128 + 128; ++t) {
      float d = tv - c_s[t];
      ls += dmf_s[t] * __expf(-DELTA * d * d - m);
    }
    red_s[r * 4 + q] = ls;
  }
  __syncthreads();
  if (j < FBLK) {
    float s = red_s[j * 4 + 0] + red_s[j * 4 + 1] +
              red_s[j * 4 + 2] + red_s[j * 4 + 3];
    sinv[j] = (s > 0.0f) ? (1.0f / s) : 0.0f;
  }

  // ---- pipelined K loop ----------------------------------------------------
  v8f acc[12];
  {
    v8f z = {0.f, 0.f, 0.f, 0.f, 0.f, 0.f, 0.f, 0.f};
#pragma unroll
    for (int i = 0; i < 12; ++i) acc[i] = z;
  }

  const int pr = j >> 2;            // 0..63  (P tile row)
  const int pk = (j & 3) * 8;       // 0,8,16,24

  for (int i = 0; i < NKIT; ++i) {
    const int k0 = i * KB;
    // (top) barrier: iteration i-1's WMMA readers of Hst[(i+1)&1] are done,
    // and Ph's previous readers are done.
    __syncthreads();

    // -- issue DMA for iteration i+1 into the other buffer
    if (i + 1 < NKIT)
      stage_hst_async(hTb, k0 + KB, &Hst[(i + 1) & 1][0], j);

    // -- P tile (branchless) + packed b128 store
    {
      float tv = tval_s[pr], m = mrow[pr], si = sinv[pr];
      v8h pv;
#pragma unroll
      for (int e = 0; e < 8; ++e) {
        int t = k0 + pk + e;
        float d = tv - c_s[t];
        float w = dmf_s[t] * __expf(-DELTA * d * d - m) * si;
        pv[e] = (_Float16)w;
      }
      *(v8h*)(&Ph[pr * PH_LD + pk]) = pv;
    }

    // -- drain current buffer's copies: async loads complete in order, so
    //    <=6 outstanding means buffer i landed while buffer i+1 is in flight
    if (i + 1 < NKIT)
      asm volatile("s_wait_asynccnt 0x6" ::: "memory");
    else
      asm volatile("s_wait_asynccnt 0x0" ::: "memory");
    __syncthreads();   // (mid) everyone's data + P tile visible

    // -- A fragment (16x32 f16, ISA layout): lane = row; elems 0-7 <- K kb..,
    //    elems 8-15 <- K kb+16.., kb = 8 for upper half-wave
    int arow = wm * 16 + (lane & 15);
    int kb   = (lane >> 4) * 8;
    const _Float16* pa = &Ph[arow * PH_LD];
    v8h a0 = *(const v8h*)(pa + kb);
    v8h a1 = *(const v8h*)(pa + kb + 16);
    v16h afrag = __builtin_shufflevector(a0, a1, 0,1,2,3,4,5,6,7,
                                                 8,9,10,11,12,13,14,15);

    // -- B fragments register-double-buffered: load nt+1 before WMMA nt so
    //    each pair of ds_load_b128 hides under the previous WMMA.
    const _Float16* Hcur = &Hst[i & 1][0];
    v16h bf = load_bfrag(Hcur, wn, 0, lane);
#pragma unroll
    for (int nt = 0; nt < 12; ++nt) {
      v16h bf_next = (nt + 1 < 12) ? load_bfrag(Hcur, wn, nt + 1, lane) : bf;
      acc[nt] = __builtin_amdgcn_wmma_f32_16x16x32_f16(
          false, afrag, false, bf, (short)0, acc[nt], false, false);
      bf = bf_next;
    }
  }

  // ---- store C/D (non-temporal: single-use stream, keep L2 for hsT) --------
  // VGPR v -> row v (+8 upper half-wave), col = lane&15
  float* outb = out + ((size_t)b * TF + fblk) * AD;
  int rbase = wm * 16 + (lane >> 4) * 8;
  int cbase = wn * 192 + (lane & 15);
#pragma unroll
  for (int nt = 0; nt < 12; ++nt) {
    int col = cbase + nt * 16;
#pragma unroll
    for (int v = 0; v < 8; ++v) {
      __builtin_nontemporal_store(acc[nt][v],
                                  &outb[(size_t)(rbase + v) * AD + col]);
    }
  }
}

// ---------------------------------------------------------------------------
// Launch
// ---------------------------------------------------------------------------
extern "C" void kernel_launch(void* const* d_in, const int* in_sizes, int n_in,
                              void* d_out, int out_size, void* d_ws, size_t ws_size,
                              hipStream_t stream) {
  const float*         hs = (const float*)d_in[0];          // (B, TT, AD)
  const float*         ds = (const float*)d_in[1];          // (B, TT)
  const unsigned char* hm = (const unsigned char*)d_in[2];  // (B, TF) bool
  const unsigned char* dm = (const unsigned char*)d_in[3];  // (B, TT) bool
  float* out = (float*)d_out;

  float*    cws = (float*)d_ws;                               // B*TT floats
  _Float16* hsT = (_Float16*)((char*)d_ws + B_ * TT * sizeof(float));

  gu_centers<<<dim3(B_), dim3(TT), 0, stream>>>(ds, cws);
  gu_transpose<<<dim3(TT / 64, AD / 64, B_), dim3(256), 0, stream>>>(hs, hsT);
  gu_main<<<dim3(TF / FBLK, B_), dim3(256), 0, stream>>>(hsT, cws, hm, dm, out);
}
